// Model_74234214744499
// MI455X (gfx1250) — matrix operand order repaired and checked by reading, not compiled, not run on hardware
//
#include <hip/hip_runtime.h>
#include <hip/hip_bf16.h>

// ---------------------------------------------------------------------------
// Persistent 2-layer GRU for MI455X (gfx1250, wave32, WMMA).
//
//  - Input path collapsed algebraically:  gi0_t = u0 * x_t + c0  (rank-1),
//    removing W_proc and W_ih0 from the sequential critical path.
//  - Weights W_hh0 / W_hh1 / W_ih1 (3 x 1536x512) held bf16 in VGPRs across
//    all 8191 steps: 288 row-tiles (16 rows x 512 K), 2 tiles per wave,
//    144 waves = 18 workgroups x 8 waves.
//  - Per step (2 grid barriers, double-buffered gate vectors):
//      phase A: gh0 = Whh0@h0, gh1 = Whh1@h1      -> barrier ->
//      h0 update (per-WG redundant, LDS) ; phase B: gi1 = Wih1@h0' -> barrier ->
//      h1 update + residual (block 0).
//  - Matvec via v_wmma_f32_16x16x32_bf16: A = h replicated across 16 M-rows
//    (built from bf16 LDS copy via ds_load_b128), B = 16-col tile of W^T.
//  - Loop-invariant update constants (u0, c0, biases) staged in LDS.
// ---------------------------------------------------------------------------

typedef __attribute__((ext_vector_type(16))) __bf16 v16bf;
typedef __attribute__((ext_vector_type(8)))  float  v8f;

#define HDIM   512
#define KDIM   512
#define G3     1536
#define NWG    18
#define TPB    256

// workspace float offsets
#define OFF_U0   0
#define OFF_C0   1536
#define OFF_G    3072            // two parity sets of 3*1536 gate values
#define GSET     4608
#define OFF_BAR  12288           // 2 unsigned ints live here

__device__ __forceinline__ float sigf(float x) {
    return 1.0f / (1.0f + __expf(-x));
}

__device__ __forceinline__ void grid_barrier(unsigned* cnt, unsigned* gen) {
    __threadfence();             // make this thread's global stores visible
    __syncthreads();
    if (threadIdx.x == 0) {
        unsigned g = __hip_atomic_load(gen, __ATOMIC_RELAXED, __HIP_MEMORY_SCOPE_AGENT);
        unsigned a = __hip_atomic_fetch_add(cnt, 1u, __ATOMIC_ACQ_REL, __HIP_MEMORY_SCOPE_AGENT);
        if (a == (unsigned)(NWG - 1)) {
            __hip_atomic_store(cnt, 0u, __ATOMIC_RELAXED, __HIP_MEMORY_SCOPE_AGENT);
            __hip_atomic_fetch_add(gen, 1u, __ATOMIC_RELEASE, __HIP_MEMORY_SCOPE_AGENT);
        } else {
            while (__hip_atomic_load(gen, __ATOMIC_ACQUIRE, __HIP_MEMORY_SCOPE_AGENT) == g) {
                __builtin_amdgcn_s_sleep(1);
            }
        }
        __threadfence();
    }
    __syncthreads();
}

// Build the replicated-h A operand for one K=32 chunk.
// 16-bit A 16x32 layout: lanes 0-15 slots 0-3 hold K=2j,2j+1; lanes 16-31 hold K=8+2j;
// slots 4-7 hold K=16.. / K=24.. respectively.  (Two contiguous 16B runs -> ds_load_b128.)
__device__ __forceinline__ v16bf build_a(const __bf16* hsb, int k0, int half) {
    v16bf a;
    unsigned* ap = (unsigned*)&a;
#pragma unroll
    for (int j = 0; j < 4; ++j) {
        ap[j]     = *(const unsigned*)(hsb + k0 + 8 * half + 2 * j);
        ap[4 + j] = *(const unsigned*)(hsb + k0 + 16 + 8 * half + 2 * j);
    }
    return a;
}

// Setup: u0 = W_ih0 @ W_proc[:,0], c0 = W_ih0 @ b_proc + b_ih0; zero barrier.
__global__ void gru_setup(const float* __restrict__ Wih0,
                          const float* __restrict__ Wproc,
                          const float* __restrict__ bproc,
                          const float* __restrict__ bih0,
                          float* __restrict__ ws) {
    int r = blockIdx.x * blockDim.x + threadIdx.x;
    if (r == 0) {
        unsigned* bar = (unsigned*)(ws + OFF_BAR);
        bar[0] = 0u;
        bar[1] = 0u;
    }
    if (r < G3) {
        float su = 0.0f, sc = 0.0f;
#pragma unroll 8
        for (int k = 0; k < 128; ++k) {
            float w = Wih0[r * 128 + k];
            su += w * Wproc[k];
            sc += w * bproc[k];
        }
        ws[OFF_U0 + r] = su;
        ws[OFF_C0 + r] = sc + bih0[r];
    }
}

__global__ __launch_bounds__(TPB, 1)
void gru_persistent(const float* __restrict__ input,
                    const float* __restrict__ residual,
                    const float* __restrict__ memory,
                    const float* __restrict__ Whh0,
                    const float* __restrict__ Wih1,
                    const float* __restrict__ Whh1,
                    const float* __restrict__ bhh0,
                    const float* __restrict__ bih1,
                    const float* __restrict__ bhh1,
                    float* __restrict__ ws,
                    const int* __restrict__ lenp,
                    float* __restrict__ out) {
    __shared__ __bf16 hs0b[HDIM];   // bf16 packed hidden states (WMMA A source)
    __shared__ __bf16 hs1b[HDIM];
    __shared__ float  hs0f[HDIM];   // f32 hidden states (update precision)
    __shared__ float  hs1f[HDIM];
    __shared__ float  resf[HDIM];   // residual accumulator (block 0 authoritative)
    // loop-invariant update constants, staged once
    __shared__ float  su0[G3];
    __shared__ float  sc0[G3];
    __shared__ float  sbh0[G3];
    __shared__ float  sbi1[G3];
    __shared__ float  sbh1[G3];

    unsigned* bar_cnt = (unsigned*)(ws + OFF_BAR);
    unsigned* bar_gen = bar_cnt + 1;

    const int tid  = threadIdx.x;
    const int lane = tid & 31;
    const int wave = tid >> 5;
    const int gw   = blockIdx.x * (TPB / 32) + wave;   // 0..143
    const int t0   = 2 * gw;                           // tiles t0, t0+1
    const int m    = t0 / 96;                          // 0: Whh0, 1: Whh1, 2: Wih1
    const int row0 = (t0 % 96) * 16;
    const int row1 = row0 + 16;
    const int n    = lane & 15;
    const int half = lane >> 4;
    const int moff = (m == 0) ? 0 : ((m == 1) ? G3 : 2 * G3);

    // Init LDS state from inputs + stage loop-invariant constants.
    for (int i = tid; i < HDIM; i += TPB) {
        float h0 = memory[i];
        float h1 = memory[HDIM + i];
        hs0f[i] = h0;  hs0b[i] = (__bf16)h0;
        hs1f[i] = h1;  hs1b[i] = (__bf16)h1;
        resf[i] = residual[i];
    }
    for (int i = tid; i < G3; i += TPB) {
        su0[i]  = ws[OFF_U0 + i];
        sc0[i]  = ws[OFF_C0 + i];
        sbh0[i] = bhh0[i];
        sbi1[i] = bih1[i];
        sbh1[i] = bhh1[i];
    }

    // Preload this wave's two weight tiles into VGPRs (bf16), B layout:
    // 32x16: lanes 0-15 hold K=0..7,16..23 (2/VGPR), lanes 16-31 hold K=8..15,24..31; col = lane&15.
    const float* Wm = (m == 0) ? Whh0 : ((m == 1) ? Whh1 : Wih1);
    v16bf wt0[16], wt1[16];
#pragma unroll 16
    for (int kc = 0; kc < 16; ++kc) {
#pragma unroll
        for (int j = 0; j < 8; ++j) {
            int K = kc * 32 + 8 * half + 2 * j + ((j >= 4) ? 8 : 0);
            wt0[kc][2 * j]     = (__bf16)Wm[(row0 + n) * KDIM + K];
            wt0[kc][2 * j + 1] = (__bf16)Wm[(row0 + n) * KDIM + K + 1];
            wt1[kc][2 * j]     = (__bf16)Wm[(row1 + n) * KDIM + K];
            wt1[kc][2 * j + 1] = (__bf16)Wm[(row1 + n) * KDIM + K + 1];
        }
    }
    __syncthreads();

    const int T = lenp[0] - 1;

    for (int t = 0; t < T; ++t) {
        float* gbase = ws + OFF_G + (t & 1) * GSET;   // parity double buffer
        float* g_gh0 = gbase;
        float* g_gh1 = gbase + G3;
        float* g_gi1 = gbase + 2 * G3;
        float* gout  = gbase + moff;

        // ---- Phase A: gh0 = Whh0 @ h0, gh1 = Whh1 @ h1 (wave-uniform branch) ----
        if (m < 2) {
            const __bf16* hsb = (m == 1) ? hs1b : hs0b;
            v8f acc0 = {};
            v8f acc1 = {};
#pragma unroll 16
            for (int kc = 0; kc < 16; ++kc) {
                v16bf a = build_a(hsb, kc * 32, half);
                acc0 = __builtin_amdgcn_wmma_f32_16x16x32_bf16(
                    false, a, false, wt0[kc], (short)0, acc0, false, false);
                acc1 = __builtin_amdgcn_wmma_f32_16x16x32_bf16(
                    false, a, false, wt1[kc], (short)0, acc1, false, false);
            }
            if (lane < 16) {
                gout[row0 + lane] = acc0[0];
                gout[row1 + lane] = acc1[0];
            }
        }
        grid_barrier(bar_cnt, bar_gen);

        // ---- h0 update (every WG redundantly, into its own LDS) ----
        {
            float x = input[t];
#pragma unroll
            for (int s = 0; s < 2; ++s) {
                int i = tid + s * TPB;
                float gr = su0[i] * x + sc0[i];
                float gz = su0[HDIM + i] * x + sc0[HDIM + i];
                float gn = su0[2 * HDIM + i] * x + sc0[2 * HDIM + i];
                float hr = g_gh0[i] + sbh0[i];
                float hz = g_gh0[HDIM + i] + sbh0[HDIM + i];
                float hn = g_gh0[2 * HDIM + i] + sbh0[2 * HDIM + i];
                float r  = sigf(gr + hr);
                float z  = sigf(gz + hz);
                float nn = tanhf(gn + r * hn);
                float hv = hs0f[i];
                float hnew = (1.0f - z) * nn + z * hv;
                hs0f[i] = hnew;
                hs0b[i] = (__bf16)hnew;
            }
        }
        __syncthreads();

        // ---- Phase B: gi1 = Wih1 @ h0_new ----
        if (m == 2) {
            v8f acc0 = {};
            v8f acc1 = {};
#pragma unroll 16
            for (int kc = 0; kc < 16; ++kc) {
                v16bf a = build_a(hs0b, kc * 32, half);
                acc0 = __builtin_amdgcn_wmma_f32_16x16x32_bf16(
                    false, a, false, wt0[kc], (short)0, acc0, false, false);
                acc1 = __builtin_amdgcn_wmma_f32_16x16x32_bf16(
                    false, a, false, wt1[kc], (short)0, acc1, false, false);
            }
            if (lane < 16) {
                gout[row0 + lane] = acc0[0];
                gout[row1 + lane] = acc1[0];
            }
        }
        grid_barrier(bar_cnt, bar_gen);

        // ---- h1 update (+ residual in block 0) ----
        {
#pragma unroll
            for (int s = 0; s < 2; ++s) {
                int i = tid + s * TPB;
                float gr = g_gi1[i] + sbi1[i];
                float gz = g_gi1[HDIM + i] + sbi1[HDIM + i];
                float gn = g_gi1[2 * HDIM + i] + sbi1[2 * HDIM + i];
                float hr = g_gh1[i] + sbh1[i];
                float hz = g_gh1[HDIM + i] + sbh1[HDIM + i];
                float hn = g_gh1[2 * HDIM + i] + sbh1[2 * HDIM + i];
                float r  = sigf(gr + hr);
                float z  = sigf(gz + hz);
                float nn = tanhf(gn + r * hn);
                float hv = hs1f[i];
                float hnew = (1.0f - z) * nn + z * hv;
                hs1f[i] = hnew;
                hs1b[i] = (__bf16)hnew;
                if (blockIdx.x == 0) {
                    resf[i] = sigf(resf[i] + hnew);
                }
            }
        }
        __syncthreads();
        // no third barrier: next step's phase A writes the other parity set,
        // and no WG can be two steps ahead (each step has two barriers).
    }

    // ---- Write outputs: residual (512), then memory = [h0, h1] (1024) ----
    if (blockIdx.x == 0) {
        for (int i = tid; i < HDIM; i += TPB) {
            out[i]            = resf[i];
            out[HDIM + i]     = hs0f[i];
            out[2 * HDIM + i] = hs1f[i];
        }
    }
}

extern "C" void kernel_launch(void* const* d_in, const int* in_sizes, int n_in,
                              void* d_out, int out_size, void* d_ws, size_t ws_size,
                              hipStream_t stream) {
    const float* input    = (const float*)d_in[0];
    const float* residual = (const float*)d_in[1];
    const float* memory   = (const float*)d_in[2];
    const float* Wproc    = (const float*)d_in[3];
    const float* bproc    = (const float*)d_in[4];
    const float* Wih0     = (const float*)d_in[5];
    const float* Whh0     = (const float*)d_in[6];
    const float* bih0     = (const float*)d_in[7];
    const float* bhh0     = (const float*)d_in[8];
    const float* Wih1     = (const float*)d_in[9];
    const float* Whh1     = (const float*)d_in[10];
    const float* bih1     = (const float*)d_in[11];
    const float* bhh1     = (const float*)d_in[12];
    const int*   lenp     = (const int*)d_in[13];

    float* ws  = (float*)d_ws;
    float* out = (float*)d_out;

    gru_setup<<<6, 256, 0, stream>>>(Wih0, Wproc, bproc, bih0, ws);
    gru_persistent<<<NWG, TPB, 0, stream>>>(input, residual, memory,
                                            Whh0, Wih1, Whh1,
                                            bhh0, bih1, bhh1,
                                            ws, lenp, out);
}
